// LSTM_Branch_32633161515020
// MI455X (gfx1250) — matrix operand order repaired and checked by reading, not compiled, not run on hardware
//
#include <hip/hip_runtime.h>

typedef __attribute__((ext_vector_type(16))) _Float16 v16h;
typedef __attribute__((ext_vector_type(8)))  _Float16 v8h;
typedef __attribute__((ext_vector_type(8)))  float    v8f;

#define T_STEPS 365
#define IN_F    3
#define HID     64
#define GATES   256      // 4*H
#define BT      16       // batch rows per block
#define HSTR    72       // LDS stride (halves) for h buffers: 144B rows, 16B aligned, bank-spread
#define GSTR    260      // LDS stride (floats) for gate buffer: de-conflicts row halves

// ---- fast activations: gfx1250 has v_tanh_f32 (TRANS pipe, co-executes with WMMA) ----
__device__ __forceinline__ float fast_tanh(float x) {
#if __has_builtin(__builtin_amdgcn_tanhf)
    return __builtin_amdgcn_tanhf(x);
#else
    float e = __expf(-2.0f * x);
    return (1.0f - e) * __builtin_amdgcn_rcpf(1.0f + e);
#endif
}
__device__ __forceinline__ float fast_sigmoid(float x) {
#if __has_builtin(__builtin_amdgcn_tanhf)
    // sigmoid(x) = 0.5 + 0.5*tanh(x/2): 1 mul + 1 tanh + 1 fma, no division
    return fmaf(0.5f, __builtin_amdgcn_tanhf(0.5f * x), 0.5f);
#else
    return __builtin_amdgcn_rcpf(1.0f + __expf(-x));
#endif
}

// A operand (16-bit 16x32): lane = M | (hi<<4); halves 0..7 -> K = ks*32 + hi*8 + j,
// halves 8..15 -> K = ks*32 + 16 + hi*8 + j. Two contiguous 16B LDS chunks.
__device__ __forceinline__ v16h load_a16(const _Float16* row, int ks, int hi) {
    union { v16h v; v8h h[2]; } u;
    const _Float16* p = row + ks * 32 + hi * 8;
    u.h[0] = *(const v8h*)(p);
    u.h[1] = *(const v8h*)(p + 16);
    return u.v;
}

// B operand (16-bit 32x16): lane&15 = N column; halves j<8 -> K = ks*32 + hi*8 + j, j>=8 -> K = ks*32 + 16 + hi*8 + (j-8).
// W is row-major [256][64] fp32, and B[k][n] = W[n][k] => two runs of 8 contiguous floats per lane.
__device__ __forceinline__ v16h load_b_w(const float* __restrict__ W, int n, int ks, int hi16) {
    union { v16h v; _Float16 e[16]; } u;
    const float* p = W + n * HID + ks * 32 + (hi16 >> 1);
#pragma unroll
    for (int j = 0; j < 16; ++j) u.e[j] = (_Float16)p[j + (j & 8)];
    return u.v;
}

__global__ __launch_bounds__(256)
void lstm2_fused_kernel(const float* __restrict__ x,
                        const float* __restrict__ w_ih0,
                        const float* __restrict__ w_hh0,
                        const float* __restrict__ b_ih0,
                        const float* __restrict__ b_hh0,
                        const float* __restrict__ w_ih1,
                        const float* __restrict__ w_hh1,
                        const float* __restrict__ b_ih1,
                        const float* __restrict__ b_hh1,
                        float* __restrict__ out,
                        int B)
{
    __shared__ __align__(16) _Float16 h1buf[BT * HSTR];   // layer-1 hidden state (f16)
    __shared__ __align__(16) _Float16 h2buf[BT * HSTR];   // layer-2 hidden state (f16)
    __shared__ __align__(16) float    gbuf[BT * GSTR];    // gate staging [16][256] (padded)
    __shared__ float xbuf[BT * IN_F];                     // x_t staging

    const int tid  = threadIdx.x;
    const int lane = tid & 31;
    const int wave = tid >> 5;          // 0..7, owns gate cols [32*wave, 32*wave+32)
    const int lh   = lane & 15;
    const int hi   = lane >> 4;         // 0/1 lane-half
    const int b0   = blockIdx.x * BT;

    // ---------------- one-time weight preload into VGPRs ----------------
    v16h bw_hh0[2][2], bw_ih1[2][2], bw_hh1[2][2];
    float wi0[2][3], bs0[2], bs1[2];
#pragma unroll
    for (int tt = 0; tt < 2; ++tt) {
        const int nt = wave * 2 + tt;
        const int n  = nt * 16 + lh;    // gate column this lane owns for tile tt
#pragma unroll
        for (int ks = 0; ks < 2; ++ks) {
            bw_hh0[tt][ks] = load_b_w(w_hh0, n, ks, hi * 16);
            bw_ih1[tt][ks] = load_b_w(w_ih1, n, ks, hi * 16);
            bw_hh1[tt][ks] = load_b_w(w_hh1, n, ks, hi * 16);
        }
        wi0[tt][0] = w_ih0[n * IN_F + 0];
        wi0[tt][1] = w_ih0[n * IN_F + 1];
        wi0[tt][2] = w_ih0[n * IN_F + 2];
        bs0[tt] = b_ih0[n] + b_hh0[n];
        bs1[tt] = b_ih1[n] + b_hh1[n];
    }

    // zero h state
    for (int i = tid; i < BT * HSTR; i += 256) {
        h1buf[i] = (_Float16)0.0f;
        h2buf[i] = (_Float16)0.0f;
    }

    // elementwise-phase ownership: each thread owns 4 consecutive hidden cols of one row
    const int erow = tid >> 4;              // 0..15
    const int ecol = (tid & 15) << 2;       // 0,4,..,60
    float c1[4] = {0.f, 0.f, 0.f, 0.f};
    float c2[4] = {0.f, 0.f, 0.f, 0.f};
    float h2o[4] = {0.f, 0.f, 0.f, 0.f};

    for (int t = 0; t < T_STEPS; ++t) {
        // ---- stage x_t ----
        if (tid < BT * IN_F) {
            int row = tid / IN_F, c = tid - row * IN_F;
            int b = b0 + row;
            xbuf[tid] = (b < B) ? x[(size_t)b * (T_STEPS * IN_F) + t * IN_F + c] : 0.0f;
        }
        __syncthreads();   // S1: x staged, h2buf(t-1) & h1buf(t-1) visible, gbuf free

        // ---------------- layer 1: gates = b + x_t Wih0^T + h1 Whh0^T ----------------
#pragma unroll
        for (int tt = 0; tt < 2; ++tt) {
            v8f acc;
#pragma unroll
            for (int r = 0; r < 8; ++r) {
                const int M = r + hi * 8;   // accumulator C-layout row
                acc[r] = fmaf(xbuf[M * IN_F + 0], wi0[tt][0],
                         fmaf(xbuf[M * IN_F + 1], wi0[tt][1],
                         fmaf(xbuf[M * IN_F + 2], wi0[tt][2], bs0[tt])));
            }
#pragma unroll
            for (int ks = 0; ks < 2; ++ks) {
                v16h a = load_a16(&h1buf[lh * HSTR], ks, hi);
                acc = __builtin_amdgcn_wmma_f32_16x16x32_f16(
                          false, a, false, bw_hh0[tt][ks], (short)0, acc, false, false);
            }
            const int ncol = (wave * 2 + tt) * 16 + lh;
#pragma unroll
            for (int r = 0; r < 8; ++r)
                gbuf[(r + hi * 8) * GSTR + ncol] = acc[r];
        }
        __syncthreads();   // S2: gates1 visible

        // ---- elementwise layer 1 ----
        {
#pragma unroll
            for (int k = 0; k < 4; ++k) {
                const int hc = ecol + k;
                const float* g = &gbuf[erow * GSTR];
                float iv = fast_sigmoid(g[hc]);
                float fv = fast_sigmoid(g[64 + hc]);
                float gv = fast_tanh(g[128 + hc]);
                float ov = fast_sigmoid(g[192 + hc]);
                c1[k] = fmaf(fv, c1[k], iv * gv);
                h1buf[erow * HSTR + hc] = (_Float16)(ov * fast_tanh(c1[k]));
            }
        }
        __syncthreads();   // S3: h1buf(t) ready, gates1 readers done -> gbuf reusable

        // ---------------- layer 2: gates = b + h1 Wih1^T + h2 Whh1^T ----------------
#pragma unroll
        for (int tt = 0; tt < 2; ++tt) {
            v8f acc;
#pragma unroll
            for (int r = 0; r < 8; ++r) acc[r] = bs1[tt];
#pragma unroll
            for (int ks = 0; ks < 2; ++ks) {
                v16h a1 = load_a16(&h1buf[lh * HSTR], ks, hi);
                acc = __builtin_amdgcn_wmma_f32_16x16x32_f16(
                          false, a1, false, bw_ih1[tt][ks], (short)0, acc, false, false);
                v16h a2 = load_a16(&h2buf[lh * HSTR], ks, hi);
                acc = __builtin_amdgcn_wmma_f32_16x16x32_f16(
                          false, a2, false, bw_hh1[tt][ks], (short)0, acc, false, false);
            }
            const int ncol = (wave * 2 + tt) * 16 + lh;
#pragma unroll
            for (int r = 0; r < 8; ++r)
                gbuf[(r + hi * 8) * GSTR + ncol] = acc[r];
        }
        __syncthreads();   // S4: gates2 visible, h2buf WMMA reads done

        // ---- elementwise layer 2 ----
        {
#pragma unroll
            for (int k = 0; k < 4; ++k) {
                const int hc = ecol + k;
                const float* g = &gbuf[erow * GSTR];
                float iv = fast_sigmoid(g[hc]);
                float fv = fast_sigmoid(g[64 + hc]);
                float gv = fast_tanh(g[128 + hc]);
                float ov = fast_sigmoid(g[192 + hc]);
                c2[k] = fmaf(fv, c2[k], iv * gv);
                h2o[k] = ov * fast_tanh(c2[k]);
                h2buf[erow * HSTR + hc] = (_Float16)h2o[k];
            }
        }
        // next iteration's S1 orders h2buf writes before layer-2 WMMA reads
    }

    // ---- final hidden state of top layer ----
    {
        const int b = b0 + erow;
        if (b < B) {
#pragma unroll
            for (int k = 0; k < 4; ++k)
                out[(size_t)b * HID + ecol + k] = h2o[k];
        }
    }
}

extern "C" void kernel_launch(void* const* d_in, const int* in_sizes, int n_in,
                              void* d_out, int out_size, void* d_ws, size_t ws_size,
                              hipStream_t stream) {
    const float* x     = (const float*)d_in[0];
    const float* w_ih0 = (const float*)d_in[1];
    const float* w_hh0 = (const float*)d_in[2];
    const float* b_ih0 = (const float*)d_in[3];
    const float* b_hh0 = (const float*)d_in[4];
    const float* w_ih1 = (const float*)d_in[5];
    const float* w_hh1 = (const float*)d_in[6];
    const float* b_ih1 = (const float*)d_in[7];
    const float* b_hh1 = (const float*)d_in[8];
    float* out = (float*)d_out;

    const int B = in_sizes[0] / (T_STEPS * IN_F);   // 4096
    const int blocks = (B + BT - 1) / BT;           // 256

    lstm2_fused_kernel<<<blocks, 256, 0, stream>>>(
        x, w_ih0, w_hh0, b_ih0, b_hh0, w_ih1, w_hh1, b_ih1, b_hh1, out, B);
}
